// SNN2_6176162971649
// MI455X (gfx1250) — matrix operand order repaired and checked by reading, not compiled, not run on hardware
//
#include <hip/hip_runtime.h>

typedef __attribute__((ext_vector_type(16))) _Float16 v16h;
typedef __attribute__((ext_vector_type(8)))  _Float16 v8h;
typedef __attribute__((ext_vector_type(2)))  _Float16 v2h;
typedef __attribute__((ext_vector_type(8)))  float    v8f;
typedef __attribute__((ext_vector_type(2)))  float    v2f;

namespace {
constexpr int kT    = 128;
constexpr int kH    = 512;
constexpr int kW    = 512;
constexpr int kTile = 32;                  // 32x32 output tile per block
constexpr int kHalo = 4;                   // 9x9 conv, pad 4
constexpr int kRows = kTile + 2 * kHalo;   // 40 LDS rows
constexpr int kCols = 48;                  // K window 24 per 16-wide subtile, padded to 32
constexpr int kPairs = kRows * kCols / 2;  // 960 float2 staging quanta
constexpr float kDecay = 0.8f;             // 1 - dt*tau_syn_inv
}

__global__ __launch_bounds__(128)
void snn2_fused(const float* __restrict__ x,
                const float* __restrict__ w,
                float* __restrict__ out)
{
    // double-buffered f16 input tile (halo + K padding), zero outside the image
    __shared__ __attribute__((aligned(32))) _Float16 xs[2][kRows * kCols];
    // banded-weight B fragments: [dy][lane][16 elems]; element e of lane l is
    // B_dy[k][n], k = 16*(e>>3)+8*(l>>4)+(e&7), n = l&15, B_dy[k][n] = w[dy][k-n] in-band else 0
    __shared__ __attribute__((aligned(32))) _Float16 wbt[9 * 32 * 16];

    const int tid  = threadIdx.x;
    const int lane = tid & 31;
    const int wid  = tid >> 5;
    const int hi   = lane >> 4;     // lane half
    const int ln15 = lane & 15;     // A-matrix row m == B-matrix col n
    const int X0 = blockIdx.x * kTile;
    const int Y0 = blockIdx.y * kTile;

    // Build the banded weight fragment table once per block.
    for (int i = tid; i < 9 * 32 * 16; i += 128) {
        int dy  = i >> 9;           // /512
        int rem = i & 511;
        int l   = rem >> 4;
        int e   = rem & 15;
        int k   = ((e >> 3) << 4) + ((l >> 4) << 3) + (e & 7);
        int d   = k - (l & 15);
        float v = (d >= 0 && d <= 8) ? w[dy * 9 + d] : 0.0f;
        wbt[i] = (_Float16)v;
    }

    // This wave's 16x16 sub-tile inside the 32x32 block tile.
    const int ty = (wid >> 1) << 4;
    const int tx = (wid & 1) << 4;

    v8f Astate = {};   // = conv(i1_t)  (conv commuted through the first EMA)
    v8f Bstate = {};   // = i2_t  (the output before update)

    for (int t = 0; t < kT; ++t) {
        const float* xt = x + (size_t)t * (kH * kW);
        _Float16* buf = xs[t & 1];

        // Stage x_t tile into LDS as f16 (2 elements per step), zero outside image.
        for (int p = tid; p < kPairs; p += 128) {
            int r  = p / (kCols / 2);
            int c  = (p - r * (kCols / 2)) * 2;
            int gy = Y0 - kHalo + r;
            int gx = X0 - kHalo + c;
            v2h pr;
            if (gy >= 0 && gy < kH && gx >= 0 && gx + 1 < kW) {
                v2f v = *(const v2f*)(xt + gy * kW + gx);        // 8B-aligned: gx even
                pr[0] = (_Float16)v[0];
                pr[1] = (_Float16)v[1];
            } else {
                float a = 0.0f, b = 0.0f;
                if (gy >= 0 && gy < kH) {
                    if (gx >= 0 && gx < kW)         a = xt[gy * kW + gx];
                    if (gx + 1 >= 0 && gx + 1 < kW) b = xt[gy * kW + gx + 1];
                }
                pr[0] = (_Float16)a;
                pr[1] = (_Float16)b;
            }
            *(v2h*)&buf[r * kCols + c] = pr;                      // 4B-aligned store
        }
        // Prefetch next frame's tile rows (global_prefetch_b8).
        if (t + 1 < kT && tid < kRows) {
            int gy = Y0 - kHalo + tid;
            if (gy >= 0 && gy < kH) {
                const float* p = xt + kH * kW + gy * kW + (X0 >= kHalo ? X0 - kHalo : 0);
                __builtin_prefetch(p, 0, 1);
                __builtin_prefetch(p + 32, 0, 1);
            }
        }
        __syncthreads();   // single barrier per step (double-buffered staging)

        // y_t = conv9x9(x_t) on this wave's 16x16 sub-tile:
        // one v_wmma_f32_16x16x32_f16 per filter row, accumulating in f32.
        v8f acc = {};
        #pragma unroll
        for (int dy = 0; dy < 9; ++dy) {
            const _Float16* arow = &buf[(ty + dy + ln15) * kCols + tx + hi * 8];
            v8h a_lo = *(const v8h*)(arow);        // K = hi*8 + 0..7
            v8h a_hi = *(const v8h*)(arow + 16);   // K = 16 + hi*8 + 0..7
            v16h a = __builtin_shufflevector(a_lo, a_hi,
                                             0, 1, 2, 3, 4, 5, 6, 7,
                                             8, 9, 10, 11, 12, 13, 14, 15);
            v16h b = *(const v16h*)&wbt[(dy * 32 + lane) * 16];
            acc = __builtin_amdgcn_wmma_f32_16x16x32_f16(
                false, a, false, b, (short)0, acc, false, false);
        }

        // out_t = i2_t ; i2 <- 0.8*i2 + A ; A <- 0.8*A + y_t
        float* ot = out + (size_t)t * (kH * kW);
        #pragma unroll
        for (int r = 0; r < 8; ++r) {
            int row = Y0 + ty + r + (hi << 3);   // C/D layout: VGPR r -> M = r + 8*(lane>=16)
            int col = X0 + tx + ln15;
            ot[row * kW + col] = Bstate[r];
            Bstate[r] = kDecay * Bstate[r] + Astate[r];
            Astate[r] = kDecay * Astate[r] + acc[r];
        }
    }
}

extern "C" void kernel_launch(void* const* d_in, const int* in_sizes, int n_in,
                              void* d_out, int out_size, void* d_ws, size_t ws_size,
                              hipStream_t stream) {
    (void)in_sizes; (void)n_in; (void)out_size; (void)d_ws; (void)ws_size;
    const float* x   = (const float*)d_in[0];
    const float* w   = (const float*)d_in[1];
    float*       out = (float*)d_out;
    dim3 grid(kW / kTile, kH / kTile, 1);
    snn2_fused<<<grid, dim3(128), 0, stream>>>(x, w, out);
}
